// BaseRNN_137438954424
// MI455X (gfx1250) — matrix operand order repaired and checked by reading, not compiled, not run on hardware
//
#include <hip/hip_runtime.h>
#include <hip/hip_bf16.h>

typedef __bf16 bf16_t;
typedef __attribute__((ext_vector_type(16))) __bf16 v16bf;
typedef __attribute__((ext_vector_type(8)))  __bf16 v8bf;
typedef __attribute__((ext_vector_type(8)))  float  v8f;

#define B_  64
#define T_  2048
#define D_  128
#define H_  128
#define G_  512   // 4*H

__device__ __forceinline__ float fast_sigmoid(float v) {
    return __builtin_amdgcn_rcpf(1.0f + __expf(-v));   // v_exp_f32 + v_rcp_f32
}
__device__ __forceinline__ float fast_tanh(float v) {
    return 2.0f * fast_sigmoid(2.0f * v) - 1.0f;
}

// Build a 16x32 bf16 A-fragment from an f32 row-major K-row.
// A layout: lane(0-15)=row M, half selects K-run; runs of 8 at k1 and k1+16.
__device__ __forceinline__ v16bf load_a_frag_f32(const float* __restrict__ row, int k1) {
    float4 f0 = *(const float4*)(row + k1);
    float4 f1 = *(const float4*)(row + k1 + 4);
    float4 f2 = *(const float4*)(row + k1 + 16);
    float4 f3 = *(const float4*)(row + k1 + 20);
    v16bf a;
    a[0]=(bf16_t)f0.x;  a[1]=(bf16_t)f0.y;  a[2]=(bf16_t)f0.z;  a[3]=(bf16_t)f0.w;
    a[4]=(bf16_t)f1.x;  a[5]=(bf16_t)f1.y;  a[6]=(bf16_t)f1.z;  a[7]=(bf16_t)f1.w;
    a[8]=(bf16_t)f2.x;  a[9]=(bf16_t)f2.y;  a[10]=(bf16_t)f2.z; a[11]=(bf16_t)f2.w;
    a[12]=(bf16_t)f3.x; a[13]=(bf16_t)f3.y; a[14]=(bf16_t)f3.z; a[15]=(bf16_t)f3.w;
    return a;
}

// Pack Wx [128,512] f32 (K x N row-major) into bf16 B-fragment layout:
// slot s = ((ntile*4 + kk)*32 + lane)*16 + e ;
// col = ntile*16 + (lane&15) ; k = kk*32 + (e>>3)*16 + (lane>>4)*8 + (e&7)
__global__ void pack_wx_kernel(const float* __restrict__ Wx, bf16_t* __restrict__ wxp) {
    int s = blockIdx.x * blockDim.x + threadIdx.x;   // 0..65535
    int e  = s & 15;
    int ln = (s >> 4) & 31;
    int kk = (s >> 9) & 3;
    int nt = s >> 11;
    wxp[s] = (bf16_t)Wx[(kk * 32 + (e >> 3) * 16 + (ln >> 4) * 8 + (e & 7)) * G_ + nt * 16 + (ln & 15)];
}

// Parallel precompute: xg[t][mtile][nt] 16x16 f32 tiles of x@Wx + b, stored in
// C-fragment layout (lane*8 contiguous floats) so the scan loads them directly
// as accumulator init. 262144 tiles, 1 wave per tile, full-chip parallel.
__global__ void __launch_bounds__(256)
xg_gemm_kernel(const float* __restrict__ x, const bf16_t* __restrict__ wxp,
               const float* __restrict__ bias, float* __restrict__ xg)
{
    const int tid = threadIdx.x, lane = tid & 31, wv = tid >> 5;
    const int id = blockIdx.x * 8 + wv;        // tile id: [t(2048)][mtile(4)][nt(32)]
    const int nt = id & 31;
    const int mtile = (id >> 5) & 3;
    const int t = id >> 7;
    const int nn = lane & 15, half = lane >> 4;

    const float* xrow = x + ((size_t)(mtile * 16 + nn) * T_ + t) * D_;
    v8f acc;
    float bv = bias[nt * 16 + nn];             // C layout: lane -> column
    #pragma unroll
    for (int v = 0; v < 8; ++v) acc[v] = bv;
    #pragma unroll
    for (int kk = 0; kk < 4; ++kk) {
        v16bf xa = load_a_frag_f32(xrow, kk * 32 + half * 8);
        v16bf bx = *(const v16bf*)(wxp + ((nt * 4 + kk) * 32 + lane) * 16);
        acc = __builtin_amdgcn_wmma_f32_16x16x32_bf16(
                  false, xa, false, bx, (short)0, acc, false, false);
    }
    *(v8f*)(xg + (size_t)id * 256 + lane * 8) = acc;
}

// Persistent scan: 4 blocks x 256 threads (8 waves), block owns 16 batch rows.
// Wave w owns n-tiles {w, 8+w, 16+w, 24+w} = all four gates for h-columns
// [w*16, w*16+16) -> LSTM elementwise stays in WMMA accumulator registers,
// c-state lives in registers, single barrier per timestep (h double-buffered).
// LDS: [0,128K) Wh bf16 B-fragments ; [128K,136K) h tile bf16, 2 buffers.
template<bool USE_XG>
__global__ void __launch_bounds__(256, 1)
lstm_recurrent_kernel(const float* __restrict__ x,
                      const bf16_t* __restrict__ wxp,
                      const float* __restrict__ xg,
                      const float* __restrict__ Wh,
                      const float* __restrict__ bias,
                      float* __restrict__ out)
{
    extern __shared__ char smem[];
    bf16_t* whp  = (bf16_t*)smem;                   // 128 KB
    bf16_t* hbuf = (bf16_t*)(smem + 128 * 1024);    // 2 x 16x128 bf16

    const int tid  = threadIdx.x;
    const int lane = tid & 31;
    const int wv   = tid >> 5;
    const int b0   = blockIdx.x * 16;
    const int nn   = lane & 15, half = lane >> 4;
    const int col  = wv * 16 + nn;                  // this lane's h-column

    // One-time: pack Wh f32 -> bf16 B-fragment layout in LDS.
    for (int s = tid; s < 32 * 4 * 32 * 16; s += 256) {
        int e = s & 15, ln = (s >> 4) & 31, kk = (s >> 9) & 3, nt = s >> 11;
        whp[s] = (bf16_t)Wh[(kk * 32 + (e >> 3) * 16 + (ln >> 4) * 8 + (e & 7)) * G_ + nt * 16 + (ln & 15)];
    }
    for (int s = tid; s < 2 * 16 * H_; s += 256) hbuf[s] = (bf16_t)0.0f;  // h0 = 0

    float bias_v[4];
    #pragma unroll
    for (int g = 0; g < 4; ++g) bias_v[g] = bias[g * 128 + col];

    float cr[8];                                    // c-state: rows v+8*half, col
    #pragma unroll
    for (int v = 0; v < 8; ++v) cr[v] = 0.0f;

    __syncthreads();

    for (int t = 0; t < T_; ++t) {
        const bf16_t* hcur  = hbuf + (t & 1) * (16 * H_);
        bf16_t*       hnext = hbuf + ((t + 1) & 1) * (16 * H_);

        // ---- h A-fragments from LDS (contiguous b128 runs) ----
        v16bf ha[4];
        const bf16_t* hrow = hcur + nn * H_;
        #pragma unroll
        for (int kk = 0; kk < 4; ++kk) {
            int k1 = kk * 32 + half * 8;
            v8bf lo = *(const v8bf*)(hrow + k1);
            v8bf hi = *(const v8bf*)(hrow + k1 + 16);
            v16bf a;
            #pragma unroll
            for (int i = 0; i < 8; ++i) { a[i] = lo[i]; a[8 + i] = hi[i]; }
            ha[kk] = a;
        }

        v16bf xa[4];
        if (!USE_XG) {
            const float* xrow = x + ((size_t)(b0 + nn) * T_ + t) * D_;
            #pragma unroll
            for (int kk = 0; kk < 4; ++kk) xa[kk] = load_a_frag_f32(xrow, kk * 32 + half * 8);
            if (t + 1 < T_) __builtin_prefetch(xrow + D_, 0, 1);
        }

        // ---- gates: acc init = xg tile (or bias + x@Wx), then + h@Wh ----
        v8f acc[4];
        #pragma unroll
        for (int g = 0; g < 4; ++g) {
            const int nt = g * 8 + wv;
            if (USE_XG) {
                const size_t tb = (((size_t)t * 4 + blockIdx.x) * 32 + nt) * 256 + lane * 8;
                acc[g] = *(const v8f*)(xg + tb);
                if (t + 1 < T_) __builtin_prefetch(xg + tb + 4 * 32 * 256, 0, 1);
            } else {
                #pragma unroll
                for (int v = 0; v < 8; ++v) acc[g][v] = bias_v[g];
                #pragma unroll
                for (int kk = 0; kk < 4; ++kk) {
                    v16bf bx = *(const v16bf*)(wxp + ((nt * 4 + kk) * 32 + lane) * 16);
                    acc[g] = __builtin_amdgcn_wmma_f32_16x16x32_bf16(
                                 false, xa[kk], false, bx, (short)0, acc[g], false, false);
                }
            }
            #pragma unroll
            for (int kk = 0; kk < 4; ++kk) {
                v16bf bh = *(const v16bf*)(whp + ((nt * 4 + kk) * 32 + lane) * 16);
                acc[g] = __builtin_amdgcn_wmma_f32_16x16x32_bf16(
                             false, ha[kk], false, bh, (short)0, acc[g], false, false);
            }
        }

        // ---- in-register LSTM update: lane=col, vgpr v -> row v+8*half ----
        float* hbase = out + ((size_t)(b0 + half * 8) * T_ + t) * H_ + col;
        #pragma unroll
        for (int v = 0; v < 8; ++v) {
            float iv = fast_sigmoid(acc[0][v]);
            float fv = fast_sigmoid(acc[1][v]);
            float gv = fast_tanh(acc[2][v]);
            float ov = fast_sigmoid(acc[3][v]);
            float cn = fv * cr[v] + iv * gv;
            float hn = ov * fast_tanh(cn);
            cr[v] = cn;
            hnext[(v + half * 8) * H_ + col] = (bf16_t)hn;
            float* p = hbase + (size_t)v * T_ * H_;           // next batch row
            p[0] = hn;                                         // hs
            p[(size_t)B_ * T_ * H_] = cn;                      // cs
        }
        __syncthreads();
    }
}

extern "C" void kernel_launch(void* const* d_in, const int* in_sizes, int n_in,
                              void* d_out, int out_size, void* d_ws, size_t ws_size,
                              hipStream_t stream) {
    const float* x   = (const float*)d_in[0];   // [64,2048,128]
    const float* Wx  = (const float*)d_in[1];   // [128,512]
    const float* Wh  = (const float*)d_in[2];   // [128,512]
    const float* b   = (const float*)d_in[3];   // [512]
    float*       out = (float*)d_out;           // hs ++ cs, each [64,2048,128] f32

    bf16_t* wxp = (bf16_t*)d_ws;                            // 128 KB packed Wx
    float*  xg  = (float*)((char*)d_ws + 256 * 1024);       // 256 MB xg tiles
    const size_t need = 256 * 1024 + (size_t)T_ * B_ * G_ * sizeof(float);

    pack_wx_kernel<<<dim3(256), dim3(256), 0, stream>>>(Wx, wxp);

    if (ws_size >= need) {
        // Full-chip parallel input projection, then lightweight scan.
        xg_gemm_kernel<<<dim3(32768), dim3(256), 0, stream>>>(x, wxp, b, xg);
        lstm_recurrent_kernel<true><<<dim3(4), dim3(256), 136 * 1024, stream>>>(
            x, wxp, xg, Wh, b, out);
    } else {
        // Fallback: fuse x-projection into the scan (only 128 KB scratch).
        lstm_recurrent_kernel<false><<<dim3(4), dim3(256), 136 * 1024, stream>>>(
            x, wxp, (const float*)nullptr, Wh, b, out);
    }
}
